// CausalSelfAttention_19559281066691
// MI455X (gfx1250) — matrix operand order repaired and checked by reading, not compiled, not run on hardware
//
#include <hip/hip_runtime.h>
#include <hip/hip_bf16.h>
#include <stdint.h>

// ---------------------------------------------------------------------------
// Problem constants (from reference)
// ---------------------------------------------------------------------------
#define T_   2048
#define C_   1024
#define H_   16
#define HD_  64
#define TOPK_ 64
#define RAMP_ 32.0f
#define MAXSPAN_ 2048.0f

typedef __attribute__((ext_vector_type(16))) __bf16 v16bf;
typedef __attribute__((ext_vector_type(8)))  float  v8f;

// POD 16-byte vector (unions with HIP's uint4 lose default ctor)
struct alignas(16) U4 { uint32_t x, y, z, w; };

union FragBF { v16bf v; uint32_t u[8]; U4 q4[2]; };
union FragF  { v8f   v; float    f[8]; };

static __device__ __forceinline__ uint32_t pack2bf(float a, float b) {
  union { __bf16 h[2]; uint32_t u; } x;
  x.h[0] = (__bf16)a; x.h[1] = (__bf16)b;
  return x.u;
}

// Low 32 bits of a generic pointer to LDS == LDS byte address (ISA 10.2:
// flat->LDS mapping truncates to addr[31:0]).
static __device__ __forceinline__ uint32_t lds_addr_of(const void* p) {
  return (uint32_t)(uintptr_t)p;
}

// ---------------------------------------------------------------------------
// Tensor Data Mover: 2-D tile Global -> LDS (D# per CDNA5 ISA 8.3/8.4)
// 6-arg builtin: (uint32x4 g0, int32x8 g1, int32x4, int32x4, int32x8, i32)
// ---------------------------------------------------------------------------
#if defined(__gfx1250__) && __has_builtin(__builtin_amdgcn_tensor_load_to_lds)
#define USE_TDM 1
typedef __attribute__((ext_vector_type(4))) unsigned int tdm_v4u;
typedef __attribute__((ext_vector_type(8))) int          tdm_v8i;
typedef __attribute__((ext_vector_type(4))) int          tdm_v4i;

static __device__ __forceinline__ void tdm_load_tile_2d(
    uint32_t lds_off, const void* gptr,
    uint32_t dim0, uint32_t dim1,        // tensor extent (elements)
    uint32_t tile0, uint32_t tile1,      // tile extent (elements)
    uint32_t stride0) {                  // dim0 line stride (elements)
  uint64_t ga = (uint64_t)(uintptr_t)gptr;
  tdm_v4u g0;
  g0.x = 1u;                                   // count=1, user descriptor
  g0.y = lds_off;                              // lds_addr (bytes)
  g0.z = (uint32_t)ga;                         // global_addr[31:0]
  g0.w = (uint32_t)(ga >> 32) | (2u << 30);    // global_addr[56:32] | type=2
  tdm_v8i g1;
  g1[0] = (int)(1u << 16);                     // data_size=1 -> 2-byte elems
  g1[1] = (int)(dim0 << 16);                   // tensor_dim0[15:0]
  g1[2] = (int)((dim0 >> 16) | (dim1 << 16));  // tensor_dim0[31:16]|dim1[15:0]
  g1[3] = (int)((dim1 >> 16) | (tile0 << 16)); // tensor_dim1[31:16]|tile_dim0
  g1[4] = (int)tile1;                          // tile_dim1 | tile_dim2=0
  g1[5] = (int)stride0;                        // tensor_dim0_stride[31:0]
  g1[6] = 0;                                   // stride0 hi | stride1 lo
  g1[7] = 0;                                   // stride1 hi
  tdm_v4i z4 = {0, 0, 0, 0};
  tdm_v8i z8 = {0, 0, 0, 0, 0, 0, 0, 0};
  __builtin_amdgcn_tensor_load_to_lds(g0, g1, z4, z4, z8, 0);
}
#endif

// ---------------------------------------------------------------------------
// fp32 -> bf16 conversion (plain)
// ---------------------------------------------------------------------------
__global__ void cvt_f32_bf16(const float* __restrict__ src,
                             __bf16* __restrict__ dst, int n) {
  int i = blockIdx.x * blockDim.x + threadIdx.x;
  if (i < n) dst[i] = (__bf16)src[i];
}

// fp32 [N][K] -> bf16 [K][N] (transposed, K-major for WMMA B fragments)
__global__ void cvt_transpose_bf16(const float* __restrict__ src,
                                   __bf16* __restrict__ dst, int N, int K) {
  int idx = blockIdx.x * blockDim.x + threadIdx.x;
  if (idx >= N * K) return;
  int n = idx % N;
  int k = idx / N;
  dst[(size_t)k * N + n] = (__bf16)src[(size_t)n * K + k];
}

// ---------------------------------------------------------------------------
// GEMM:  C[M,N] = A[M,K] @ Bt[K,N]   (bf16 in, fp32 accum/out)
// Workgroup: 256 threads = 8 waves (4M x 2N); block tile 128x128, K step 32.
// Wave tile 32x64 -> 2 A-frags x 4 B-frags -> 8 WMMA per K-step.
// Double-buffered TDM pipeline: wave0 streams A tiles, wave1 streams B tiles;
// each waits only its own TENSORcnt (<=1 steady state), so DMA for tile i+1
// overlaps the 8 WMMAs of tile i.
// ---------------------------------------------------------------------------
#define GT_BM 128
#define GT_BN 128
#define GT_K  32

__global__ __launch_bounds__(256)
void gemm_bf16_wmma(const __bf16* __restrict__ A, const __bf16* __restrict__ Bt,
                    float* __restrict__ Cout, int M, int N, int K) {
  __shared__ __bf16 As[2][GT_BM * GT_K];   // 2 x 8 KB, [row][k]
  __shared__ __bf16 Bs[2][GT_K * GT_BN];   // 2 x 8 KB, [k][n]

  const int tid  = threadIdx.x;
  const int wave = tid >> 5;
  const int lane = tid & 31;
  const int wm   = wave >> 1;           // 0..3 : M sub-tile (32 rows)
  const int wn   = wave & 1;            // 0..1 : N sub-tile (64 cols)
  const int m0   = blockIdx.x * GT_BM;
  const int n0   = blockIdx.y * GT_BN;

  const int lrow  = lane & 15;
  const int khalf = (lane >> 4) << 3;   // 0 or 8

  FragF acc[2][4];
#pragma unroll
  for (int mi = 0; mi < 2; ++mi)
#pragma unroll
    for (int nj = 0; nj < 4; ++nj)
#pragma unroll
      for (int r = 0; r < 8; ++r) acc[mi][nj].f[r] = 0.0f;

  // fragment gather (b128 A reads, 16-bit B reads) + 8 WMMAs on one staged buffer
  auto compute_step = [&](const __bf16* AsP, const __bf16* BsP) {
    FragBF af[2], bf[4];
#pragma unroll
    for (int mi = 0; mi < 2; ++mi) {
      const U4* Ar = (const U4*)AsP + (size_t)(wm * 32 + mi * 16 + lrow) * 4;
      af[mi].q4[0] = Ar[khalf >> 3];        // K = khalf..khalf+7 pairs
      af[mi].q4[1] = Ar[2 + (khalf >> 3)];  // K = 16+khalf..
    }
#pragma unroll
    for (int nj = 0; nj < 4; ++nj) {
      const int nb = wn * 64 + nj * 16;     // N base of this frag
      const __bf16* Bc = BsP + nb + lrow;   // lane = N, element s = K (table order)
#pragma unroll
      for (int s = 0; s < 16; ++s) {
        const int kk = (s < 8) ? (khalf + s) : (16 + khalf + s - 8);
        bf[nj].v[s] = Bc[kk * GT_BN];
      }
    }
#pragma unroll
    for (int mi = 0; mi < 2; ++mi)
#pragma unroll
      for (int nj = 0; nj < 4; ++nj)
        acc[mi][nj].v = __builtin_amdgcn_wmma_f32_16x16x32_bf16(
            false, af[mi].v, false, bf[nj].v, (short)0, acc[mi][nj].v,
            false, false);
  };

  const int nsteps = K / GT_K;

#if USE_TDM
  // ---- prologue: stage tile 0 (A stream on wave0, B stream on wave1)
  if (wave == 0) {
    tdm_load_tile_2d(lds_addr_of(As[0]), A + (size_t)m0 * K,
                     (uint32_t)K, (uint32_t)M, GT_K, GT_BM, (uint32_t)K);
  } else if (wave == 1) {
    tdm_load_tile_2d(lds_addr_of(Bs[0]), Bt + n0,
                     (uint32_t)N, (uint32_t)K, GT_BN, GT_K, (uint32_t)N);
  }
  for (int i = 0; i < nsteps; ++i) {
    const int buf = i & 1;
    if (i + 1 < nsteps) {
      const int k1 = (i + 1) * GT_K;
      if (wave == 0) {
        tdm_load_tile_2d(lds_addr_of(As[buf ^ 1]), A + (size_t)m0 * K + k1,
                         (uint32_t)K, (uint32_t)M, GT_K, GT_BM, (uint32_t)K);
        __builtin_amdgcn_s_wait_tensorcnt(1);   // tile i's A-load done
      } else if (wave == 1) {
        tdm_load_tile_2d(lds_addr_of(Bs[buf ^ 1]), Bt + (size_t)k1 * N + n0,
                         (uint32_t)N, (uint32_t)K, GT_BN, GT_K, (uint32_t)N);
        __builtin_amdgcn_s_wait_tensorcnt(1);   // tile i's B-load done
      }
    } else {
      if (wave < 2) __builtin_amdgcn_s_wait_tensorcnt(0);
    }
    __syncthreads();                     // staged buffer visible to all waves
    compute_step(As[buf], Bs[buf]);
    __syncthreads();                     // all reads done before buf is reused
  }
#else
  const U4* Ag4 = (const U4*)A;     // 8 bf16 per U4
  const U4* Bg4 = (const U4*)Bt;
  const int Ku4 = K >> 3;           // A row stride in U4
  const int Nu4 = N >> 3;           // Bt row stride in U4
  for (int i = 0; i < nsteps; ++i) {
    const int k0 = i * GT_K;
    U4* As4 = (U4*)As[0];               // 512 U4
    U4* Bs4 = (U4*)Bs[0];               // 512 U4
#pragma unroll
    for (int j = 0; j < 2; ++j) {
      int idx = tid + j * 256;
      int r = idx >> 2, c = idx & 3;    // A: 128 rows x 4 U4
      As4[idx] = Ag4[(size_t)(m0 + r) * Ku4 + (k0 >> 3) + c];
    }
#pragma unroll
    for (int j = 0; j < 2; ++j) {
      int idx = tid + j * 256;
      int r = idx >> 4, c = idx & 15;   // B: 32 rows x 16 U4
      Bs4[idx] = Bg4[(size_t)(k0 + r) * Nu4 + (n0 >> 3) + c];
    }
    if (k0 + GT_K < K)
      __builtin_prefetch(&Ag4[(size_t)(m0 + (tid >> 1)) * Ku4 + ((k0 + GT_K) >> 3)], 0, 1);
    __syncthreads();
    compute_step(As[0], Bs[0]);
    __syncthreads();
  }
#endif

  // ---- write: D layout lane -> N=lane&15, VGPR r -> M=r(+8)
  const int nlo = lane & 15;
  const int mhi = (lane >> 4) << 3;
#pragma unroll
  for (int mi = 0; mi < 2; ++mi)
#pragma unroll
    for (int nj = 0; nj < 4; ++nj)
#pragma unroll
      for (int r = 0; r < 8; ++r) {
        int mq = m0 + wm * 32 + mi * 16 + mhi + r;
        int nq = n0 + wn * 64 + nj * 16 + nlo;
        Cout[(size_t)mq * N + nq] = acc[mi][nj].f[r];
      }
}

// ---------------------------------------------------------------------------
// RoPE + head split:  qkv[T,3C] fp32 ->
//   qb [H][T][HD] bf16, kbT [H][HD][T] bf16 (dim-major!), vb [H][T][HD] bf16
// ---------------------------------------------------------------------------
__global__ void rope_split_kernel(const float* __restrict__ qkv,
                                  __bf16* __restrict__ qb,
                                  __bf16* __restrict__ kbT,
                                  __bf16* __restrict__ vb) {
  int idx = blockIdx.x * blockDim.x + threadIdx.x;
  if (idx >= T_ * H_ * (HD_ / 2)) return;
  int d = idx & 31;
  int h = (idx >> 5) & (H_ - 1);
  int t = idx >> 9;

  const float* base = qkv + (size_t)t * 3 * C_;
  float q1 = base[h * HD_ + d],          q2 = base[h * HD_ + d + 32];
  float k1 = base[C_ + h * HD_ + d],     k2 = base[C_ + h * HD_ + d + 32];
  float v1 = base[2 * C_ + h * HD_ + d], v2 = base[2 * C_ + h * HD_ + d + 32];

  float inv = powf(10000.0f, -(2.0f * (float)d) / (float)HD_);
  float ang = (float)t * inv;
  float c = __cosf(ang), s = __sinf(ang);

  size_t off = ((size_t)h * T_ + t) * HD_;
  qb[off + d]      = (__bf16)(q1 * c - q2 * s);
  qb[off + d + 32] = (__bf16)(q2 * c + q1 * s);
  vb[off + d]      = (__bf16)v1;
  vb[off + d + 32] = (__bf16)v2;
  size_t offT = ((size_t)h * HD_ + d) * T_ + t;       // [h][d][t]
  kbT[offT]                 = (__bf16)(k1 * c - k2 * s);
  kbT[offT + (size_t)32 * T_] = (__bf16)(k2 * c + k1 * s);
}

// ---------------------------------------------------------------------------
// Attention: one workgroup = (head h, 16-query tile). 16x2048 fp32 score
// strip lives in LDS (128 KB of the WGP's 320 KB).
// ---------------------------------------------------------------------------
__global__ __launch_bounds__(256)
void attn_kernel(const __bf16* __restrict__ qb, const __bf16* __restrict__ kbT,
                 const __bf16* __restrict__ vb,
                 const float* __restrict__ span_params,
                 float* __restrict__ yattn) {
  extern __shared__ float smem[];
  float* sc    = smem;             // [16][T_]
  float* ypart = smem + 16 * T_;   // [4][256] partial PV tiles

  const int q0   = blockIdx.x * 16;
  const int h    = blockIdx.y;
  const int tid  = threadIdx.x;
  const int wave = tid >> 5;
  const int lane = tid & 31;

  const int L       = q0 + 16;
  const int nkTiles = (L + 127) >> 7;
  const int Lcov    = nkTiles * 128;
  const float span  = MAXSPAN_ * fminf(fmaxf(span_params[h], 0.0f), 1.0f);

  const __bf16* qh  = qb  + (size_t)h * T_ * HD_;
  const __bf16* khT = kbT + (size_t)h * HD_ * T_;
  const __bf16* vh  = vb  + (size_t)h * T_ * HD_;

  // ---- Q fragments (A layout), all b128 loads
  const int arow  = lane & 15;
  const int khalf = (lane >> 4) << 3;
  FragBF qf[2];
  {
    const U4* qg4 = (const U4*)(qh + (size_t)(q0 + arow) * HD_);
#pragma unroll
    for (int c = 0; c < 2; ++c) {
      qf[c].q4[0] = qg4[(c * 32 + khalf) >> 3];
      qf[c].q4[1] = qg4[((c * 32 + khalf) >> 3) + 2];
    }
  }

  // ================= Phase 1: S = Q K^T / 8 =================
  for (int tile = 0; tile < nkTiles; ++tile) {
    const int kbase = tile * 128 + wave * 16;
    FragF accs;
#pragma unroll
    for (int r = 0; r < 8; ++r) accs.f[r] = 0.0f;
#pragma unroll
    for (int c = 0; c < 2; ++c) {
      // B frag: lane = key (N), element s = dim (K) in table order
      FragBF bk;
#pragma unroll
      for (int s = 0; s < 16; ++s) {
        const int kk = (s < 8) ? (khalf + s) : (16 + khalf + s - 8);
        bk.v[s] = khT[(size_t)(c * 32 + kk) * T_ + kbase + arow];
      }
      accs.v = __builtin_amdgcn_wmma_f32_16x16x32_bf16(
          false, qf[c].v, false, bk.v, (short)0, accs.v, false, false);
    }
    const int n   = lane & 15;
    const int mhi = (lane >> 4) << 3;
#pragma unroll
    for (int r = 0; r < 8; ++r) {
      int m   = mhi + r;
      int key = kbase + n;
      float s = accs.f[r] * 0.125f;
      if (key > q0 + m) s = -3.4e38f;
      sc[m * T_ + key] = s;
    }
  }
  __syncthreads();

  // ================= Phase 2: top-64 / softmax / span =================
  for (int rr = 0; rr < 2; ++rr) {
    const int m  = wave * 2 + rr;
    const int Lr = q0 + m + 1;
    float* rowp  = sc + m * T_;

    float mx = -3.4e38f, mn = 3.4e38f;
    for (int k = lane; k < Lr; k += 32) {
      float v = rowp[k];
      mx = fmaxf(mx, v); mn = fminf(mn, v);
    }
#pragma unroll
    for (int o = 16; o >= 1; o >>= 1) {
      mx = fmaxf(mx, __shfl_xor(mx, o, 32));
      mn = fminf(mn, __shfl_xor(mn, o, 32));
    }

    float thresh = mn;
    if (Lr > TOPK_) {
      float lo = mn, hi = mx;
      for (int it = 0; it < 22; ++it) {
        float mid = 0.5f * (lo + hi);
        int cnt = 0;
        for (int k = lane; k < Lr; k += 32) cnt += (rowp[k] >= mid) ? 1 : 0;
#pragma unroll
        for (int o = 16; o >= 1; o >>= 1) cnt += __shfl_xor(cnt, o, 32);
        if (cnt >= TOPK_) lo = mid; else hi = mid;
      }
      thresh = lo;
    }

    float E = 0.0f, Em = 0.0f;
    for (int k = lane; k < Lr; k += 32) {
      float v = rowp[k];
      float e = (v >= thresh) ? __expf(v - mx) : 0.0f;
      float dist = (float)(q0 + m - k);
      float msk  = fminf(fmaxf((RAMP_ + span - dist) / RAMP_, 0.0f), 1.0f);
      E += e; Em += e * msk;
    }
#pragma unroll
    for (int o = 16; o >= 1; o >>= 1) {
      E  += __shfl_xor(E,  o, 32);
      Em += __shfl_xor(Em, o, 32);
    }
    float denom = Em + 1e-8f * E;
    float invd  = (denom > 0.0f) ? 1.0f / denom : 0.0f;

    for (int k = lane; k < Lcov; k += 32) {
      float w = 0.0f;
      if (k < Lr) {
        float v = rowp[k];
        float e = (v >= thresh) ? __expf(v - mx) : 0.0f;
        float dist = (float)(q0 + m - k);
        float msk  = fminf(fmaxf((RAMP_ + span - dist) / RAMP_, 0.0f), 1.0f);
        w = e * msk * invd;
      }
      rowp[k] = w;
    }
  }
  __syncthreads();

  // ================= Phase 3: Y = P @ V =================
  const int dchunk = (wave & 3) * 16;
  const int kpar   = wave >> 2;
  FragF acc;
#pragma unroll
  for (int r = 0; r < 8; ++r) acc.f[r] = 0.0f;

  const int nkc = nkTiles * 4;
  for (int kc = kpar; kc < nkc; kc += 2) {
    const int kb0 = kc * 32;
    FragBF pa, vf;
#pragma unroll
    for (int i = 0; i < 4; ++i) {
      float f0 = sc[arow * T_ + kb0 + khalf + 2 * i];
      float f1 = sc[arow * T_ + kb0 + khalf + 2 * i + 1];
      pa.u[i] = pack2bf(f0, f1);
      float g0 = sc[arow * T_ + kb0 + 16 + khalf + 2 * i];
      float g1 = sc[arow * T_ + kb0 + 16 + khalf + 2 * i + 1];
      pa.u[i + 4] = pack2bf(g0, g1);
    }
#pragma unroll
    for (int s = 0; s < 16; ++s) {
      const int kk = (s < 8) ? (khalf + s) : (16 + khalf + s - 8);
      vf.v[s] = vh[(size_t)(kb0 + kk) * HD_ + dchunk + arow];
    }
    acc.v = __builtin_amdgcn_wmma_f32_16x16x32_bf16(
        false, pa.v, false, vf.v, (short)0, acc.v, false, false);
  }

  if (wave >= 4) {
    float* p = ypart + (wave - 4) * 256 + lane * 8;
#pragma unroll
    for (int r = 0; r < 8; ++r) p[r] = acc.f[r];
  }
  __syncthreads();
  if (wave < 4) {
    const float* p = ypart + wave * 256 + lane * 8;
    const int n   = lane & 15;
    const int mhi = (lane >> 4) << 3;
#pragma unroll
    for (int r = 0; r < 8; ++r) {
      float v = acc.f[r] + p[r];
      int mq  = mhi + r;
      yattn[(size_t)(q0 + mq) * C_ + h * HD_ + dchunk + n] = v;
    }
  }
}

// ---------------------------------------------------------------------------
// y = yattn * silu(gate_pre) -> bf16
// ---------------------------------------------------------------------------
__global__ void gate_mul_kernel(const float* __restrict__ ya,
                                const float* __restrict__ gp,
                                __bf16* __restrict__ yg, int n) {
  int i = blockIdx.x * blockDim.x + threadIdx.x;
  if (i < n) {
    float g = gp[i];
    float s = g / (1.0f + __expf(-g));
    yg[i] = (__bf16)(ya[i] * s);
  }
}

// ---------------------------------------------------------------------------
// Host launcher
// ---------------------------------------------------------------------------
extern "C" void kernel_launch(void* const* d_in, const int* in_sizes, int n_in,
                              void* d_out, int out_size, void* d_ws, size_t ws_size,
                              hipStream_t stream) {
  const float* x      = (const float*)d_in[0];   // [T, C]
  const float* w_attn = (const float*)d_in[1];   // [3C, C]
  const float* w_proj = (const float*)d_in[2];   // [C, C]
  const float* w_gate = (const float*)d_in[3];   // [C, C]
  const float* span   = (const float*)d_in[4];   // [H]
  (void)in_sizes; (void)n_in; (void)out_size; (void)ws_size;

  char* ws = (char*)d_ws;
  __bf16* xb   = (__bf16*)ws;                 ws += (size_t)T_ * C_ * 2;
  __bf16* wab  = (__bf16*)ws;                 ws += (size_t)3 * C_ * C_ * 2;  // [K][3C]
  __bf16* wgb  = (__bf16*)ws;                 ws += (size_t)C_ * C_ * 2;      // [K][C]
  __bf16* wpb  = (__bf16*)ws;                 ws += (size_t)C_ * C_ * 2;      // [K][C]
  float*  qkv  = (float*)ws;                  ws += (size_t)T_ * 3 * C_ * 4;
  float*  gpre = (float*)ws;                  ws += (size_t)T_ * C_ * 4;
  __bf16* qbuf = (__bf16*)ws;                 ws += (size_t)H_ * T_ * HD_ * 2;
  __bf16* kbufT= (__bf16*)ws;                 ws += (size_t)H_ * HD_ * T_ * 2;
  __bf16* vbuf = (__bf16*)ws;                 ws += (size_t)H_ * T_ * HD_ * 2;
  float*  yatt = (float*)ws;                  ws += (size_t)T_ * C_ * 4;
  __bf16* yg   = (__bf16*)ws;                 ws += (size_t)T_ * C_ * 2;

  const int thr = 256;
  // 1) conversions: activations plain, weights transposed to K-major
  cvt_f32_bf16<<<(T_ * C_ + thr - 1) / thr, thr, 0, stream>>>(x, xb, T_ * C_);
  cvt_transpose_bf16<<<(3 * C_ * C_ + thr - 1) / thr, thr, 0, stream>>>(
      w_attn, wab, 3 * C_, C_);
  cvt_transpose_bf16<<<(C_ * C_ + thr - 1) / thr, thr, 0, stream>>>(
      w_gate, wgb, C_, C_);
  cvt_transpose_bf16<<<(C_ * C_ + thr - 1) / thr, thr, 0, stream>>>(
      w_proj, wpb, C_, C_);

  // 2) qkv = x @ w_attn^T   [2048, 3072]
  gemm_bf16_wmma<<<dim3(T_ / GT_BM, 3 * C_ / GT_BN), thr, 0, stream>>>(
      xb, wab, qkv, T_, 3 * C_, C_);
  // 3) gate_pre = x @ w_gate^T  [2048, 1024]
  gemm_bf16_wmma<<<dim3(T_ / GT_BM, C_ / GT_BN), thr, 0, stream>>>(
      xb, wgb, gpre, T_, C_, C_);

  // 4) RoPE + head split (K stored dim-major for QK^T B-fragments)
  rope_split_kernel<<<(T_ * H_ * (HD_ / 2) + thr - 1) / thr, thr, 0, stream>>>(
      qkv, qbuf, kbufT, vbuf);

  // 5) attention (dynamic LDS: 16x2048 fp32 scores + partials = 132 KB)
  const size_t attn_lds = (size_t)16 * T_ * 4 + 4 * 256 * 4;
  (void)hipFuncSetAttribute(reinterpret_cast<const void*>(attn_kernel),
                            hipFuncAttributeMaxDynamicSharedMemorySize,
                            (int)attn_lds);
  attn_kernel<<<dim3(T_ / 16, H_), thr, attn_lds, stream>>>(
      qbuf, kbufT, vbuf, span, yatt);

  // 6) gating
  gate_mul_kernel<<<(T_ * C_ + thr - 1) / thr, thr, 0, stream>>>(
      yatt, gpre, yg, T_ * C_);

  // 7) out = (yatt*silu) @ w_proj^T -> d_out
  gemm_bf16_wmma<<<dim3(T_ / GT_BM, C_ / GT_BN), thr, 0, stream>>>(
      yg, wpb, (float*)d_out, T_, C_, C_);
}
